// DescriptorLoss_31215822308011
// MI455X (gfx1250) — compile-verified, matrix operand order checked by reading, not run on hardware
//
#include <hip/hip_runtime.h>
#include <hip/hip_bf16.h>
#include <math.h>

// ---------------------------------------------------------------------------
// Problem constants (from reference): b=2, d=256, hc=60, wc=80
// ---------------------------------------------------------------------------
#define BATCH     2
#define DCH       256
#define HC        60
#define WCC       80
#define NCELL     (HC * WCC)          // 4800
#define TILES     (NCELL / 16)        // 300 (exact)
#define KCHUNKS   (DCH / 32)          // 8
#define NWAVES    8                   // 256-thread blocks = 8 wave32
#define LAMBDA_D  250.0f
#define POS_M     1.0f
#define NEG_M     0.2f
#define EPSN      1e-12f
#define DIST2     56.25f              // 7.5^2

typedef _Float16 v16h __attribute__((ext_vector_type(16)));
typedef _Float16 v8h  __attribute__((ext_vector_type(8)));
typedef float    v8f  __attribute__((ext_vector_type(8)));

// ---------------------------------------------------------------------------
// Fragment load for V_WMMA_F32_16X16X32_F16 (16-bit A 16x32 layout, ISA 7.12.2):
//   lanes 0-15 : row M = lane,     K = 0..7  (V0..3) and K = 16..23 (V4..7)
//   lanes 16-31: row M = lane-16,  K = 8..15 (V0..3) and K = 24..31 (V4..7)
// rowBase points at element (row, kc*32) of a row-major [N][256] f16 array.
// Both A and B come from K-contiguous row-major storage (C = dn * wn^T), so
// the same loader serves both operands.
// ---------------------------------------------------------------------------
__device__ __forceinline__ v16h load_frag16x32(const _Float16* rowBase, int hi) {
    const v8h* p0 = reinterpret_cast<const v8h*>(rowBase + hi * 8);        // K 0-7 / 8-15
    const v8h* p1 = reinterpret_cast<const v8h*>(rowBase + 16 + hi * 8);   // K 16-23 / 24-31
    v8h a = *p0;
    v8h b = *p1;
    return __builtin_shufflevector(a, b, 0,1,2,3,4,5,6,7,8,9,10,11,12,13,14,15);
}

// Unconditional speculative prefetch of a 512-byte descriptor row
// (per-lane; covers next tile's 16 rows). Caller clamps the tile index.
__device__ __forceinline__ void prefetch_row(const _Float16* rowBase) {
    const char* p = reinterpret_cast<const char*>(rowBase);
    __builtin_prefetch(p, 0, 3);
    __builtin_prefetch(p + 256, 0, 3);
}

__device__ __forceinline__ int clamp_tile(int t) {
    return t < (TILES - 1) ? t : (TILES - 1);
}

// ---------------------------------------------------------------------------
// Kernel 1: normalize descriptors -> f16 (row-major [b][n][256]), compute
// homography-warped cell centers, zero the reduction buffers.
// desc layout is [b][c][i]: consecutive threads i give coalesced loads.
// ---------------------------------------------------------------------------
__global__ void k_prep(const float* __restrict__ desc,
                       const float* __restrict__ wdesc,
                       const float* __restrict__ Hm,
                       _Float16* __restrict__ dn,
                       _Float16* __restrict__ wn,
                       float* __restrict__ warp,
                       float* __restrict__ rbuf,
                       float* __restrict__ cbuf,
                       float* __restrict__ out) {
    int idx = blockIdx.x * blockDim.x + threadIdx.x;
    if (idx >= BATCH * NCELL) return;
    int batch = idx / NCELL;
    int i     = idx - batch * NCELL;

    // --- normalize desc ---
    float ss = 0.f;
    #pragma unroll 4
    for (int c = 0; c < DCH; ++c) {
        float v = desc[(batch * DCH + c) * NCELL + i];
        ss += v * v;
    }
    float inv = 1.0f / fmaxf(sqrtf(ss), EPSN);
    #pragma unroll 4
    for (int c = 0; c < DCH; ++c) {
        float v = desc[(batch * DCH + c) * NCELL + i];
        dn[(size_t)(batch * NCELL + i) * DCH + c] = (_Float16)(v * inv);
    }

    // --- normalize wrap_desc ---
    ss = 0.f;
    #pragma unroll 4
    for (int c = 0; c < DCH; ++c) {
        float v = wdesc[(batch * DCH + c) * NCELL + i];
        ss += v * v;
    }
    inv = 1.0f / fmaxf(sqrtf(ss), EPSN);
    #pragma unroll 4
    for (int c = 0; c < DCH; ++c) {
        float v = wdesc[(batch * DCH + c) * NCELL + i];
        wn[(size_t)(batch * NCELL + i) * DCH + c] = (_Float16)(v * inv);
    }

    // --- homography-warped cell center (row, col) ---
    int row = i / WCC, col = i - row * WCC;
    float cx = col * 8.0f + 4.0f;     // x
    float cy = row * 8.0f + 4.0f;     // y
    const float* H = Hm + batch * 9;
    float wx = H[0] * cx + H[1] * cy + H[2];
    float wy = H[3] * cx + H[4] * cy + H[5];
    float ww = H[6] * cx + H[7] * cy + H[8];
    warp[(batch * NCELL + i) * 2 + 0] = wy / ww;   // row
    warp[(batch * NCELL + i) * 2 + 1] = wx / ww;   // col

    rbuf[idx] = 0.f;
    cbuf[idx] = 0.f;
    if (idx == 0) out[0] = 0.f;
}

// ---------------------------------------------------------------------------
// Kernel 2 (pass A): row norms. Each wave owns one 16-row M-tile, keeps all 8
// A fragments in VGPRs, and sweeps N-tiles accumulating sum_k relu(C)^2 per row.
// launch_bounds(256,1) unlocks the VGPR budget so A (64) + full-tile B (64)
// stay in registers; the sched_barrier pins all 16 B loads in one clause
// before the WMMA chain; K-chunks alternate between two accumulators.
// ---------------------------------------------------------------------------
__global__ void __launch_bounds__(256, 1)
k_rownorm(const _Float16* __restrict__ dn,
          const _Float16* __restrict__ wn,
          float* __restrict__ rbuf) {
    int lane = threadIdx.x & 31;
    int wave = threadIdx.x >> 5;
    int m0   = blockIdx.x * 16;
    int batch = blockIdx.y;
    int hi   = lane >> 4;
    int lm   = lane & 15;

    const _Float16* arow = dn + (size_t)(batch * NCELL + m0 + lm) * DCH;

    v16h afrag[KCHUNKS];
    #pragma unroll
    for (int kc = 0; kc < KCHUNKS; ++kc)
        afrag[kc] = load_frag16x32(arow + kc * 32, hi);

    float rs[8];
    #pragma unroll
    for (int v = 0; v < 8; ++v) rs[v] = 0.f;

    const _Float16* wbase = wn + (size_t)batch * NCELL * DCH;

    for (int nt = wave; nt < TILES; nt += NWAVES) {
        const _Float16* brow = wbase + (size_t)(nt * 16 + lm) * DCH;
        prefetch_row(wbase + (size_t)(clamp_tile(nt + NWAVES) * 16 + lm) * DCH);

        // one load clause for the whole tile's B operand (kept live together)
        v16h bf[KCHUNKS];
        #pragma unroll
        for (int kc = 0; kc < KCHUNKS; ++kc)
            bf[kc] = load_frag16x32(brow + kc * 32, hi);

        __builtin_amdgcn_sched_barrier(0);   // all loads issue before any WMMA

        v8f acc0 = {}, acc1 = {};
        #pragma unroll
        for (int kc = 0; kc < KCHUNKS; kc += 2) {
            acc0 = __builtin_amdgcn_wmma_f32_16x16x32_f16(
                false, afrag[kc],     false, bf[kc],     (short)0, acc0, false, false);
            acc1 = __builtin_amdgcn_wmma_f32_16x16x32_f16(
                false, afrag[kc + 1], false, bf[kc + 1], (short)0, acc1, false, false);
        }
        #pragma unroll
        for (int v = 0; v < 8; ++v) {
            float f = fmaxf(acc0[v] + acc1[v], 0.f);     // relu
            rs[v] += f * f;
        }
    }

    // reduce over the 16 lanes of each half (same row set per half)
    #pragma unroll
    for (int v = 0; v < 8; ++v) {
        rs[v] += __shfl_xor(rs[v], 1);
        rs[v] += __shfl_xor(rs[v], 2);
        rs[v] += __shfl_xor(rs[v], 4);
        rs[v] += __shfl_xor(rs[v], 8);
    }
    if (lm == 0) {
        #pragma unroll
        for (int v = 0; v < 8; ++v)
            atomicAdd(&rbuf[batch * NCELL + m0 + v + 8 * hi], rs[v]);
    }
}

// ---------------------------------------------------------------------------
// Kernel 3: finalize a sum-of-squares buffer into 1 / max(sqrt(x), eps)
// ---------------------------------------------------------------------------
__global__ void k_finalize_inv(float* __restrict__ p, int count) {
    int idx = blockIdx.x * blockDim.x + threadIdx.x;
    if (idx >= count) return;
    p[idx] = 1.0f / fmaxf(sqrtf(p[idx]), EPSN);
}

// ---------------------------------------------------------------------------
// Kernel 4 (pass B): column norms of the row-normalized matrix. Each wave owns
// one 16-col N-tile (B fragments resident), sweeps M-tiles.
// ---------------------------------------------------------------------------
__global__ void __launch_bounds__(256, 1)
k_colnorm(const _Float16* __restrict__ dn,
          const _Float16* __restrict__ wn,
          const float* __restrict__ invr,
          float* __restrict__ cbuf) {
    int lane = threadIdx.x & 31;
    int wave = threadIdx.x >> 5;
    int n0   = blockIdx.x * 16;
    int batch = blockIdx.y;
    int hi   = lane >> 4;
    int lm   = lane & 15;

    const _Float16* brow = wn + (size_t)(batch * NCELL + n0 + lm) * DCH;
    v16h bfrag[KCHUNKS];
    #pragma unroll
    for (int kc = 0; kc < KCHUNKS; ++kc)
        bfrag[kc] = load_frag16x32(brow + kc * 32, hi);

    float cs = 0.f;  // partial column sum for column n0 + lm (this lane's half)

    const _Float16* dbase = dn + (size_t)batch * NCELL * DCH;

    for (int mt = wave; mt < TILES; mt += NWAVES) {
        const _Float16* arow = dbase + (size_t)(mt * 16 + lm) * DCH;
        prefetch_row(dbase + (size_t)(clamp_tile(mt + NWAVES) * 16 + lm) * DCH);

        v16h af[KCHUNKS];
        #pragma unroll
        for (int kc = 0; kc < KCHUNKS; ++kc)
            af[kc] = load_frag16x32(arow + kc * 32, hi);

        __builtin_amdgcn_sched_barrier(0);   // all loads issue before any WMMA

        v8f acc0 = {}, acc1 = {};
        #pragma unroll
        for (int kc = 0; kc < KCHUNKS; kc += 2) {
            acc0 = __builtin_amdgcn_wmma_f32_16x16x32_f16(
                false, af[kc],     false, bfrag[kc],     (short)0, acc0, false, false);
            acc1 = __builtin_amdgcn_wmma_f32_16x16x32_f16(
                false, af[kc + 1], false, bfrag[kc + 1], (short)0, acc1, false, false);
        }

        float rl = invr[batch * NCELL + mt * 16 + lm]; // lane lm holds row lm's inv-norm
        #pragma unroll
        for (int v = 0; v < 8; ++v) {
            float ir = __shfl(rl, v + 8 * hi);         // inv-norm of row v+8*hi
            float f  = fmaxf(acc0[v] + acc1[v], 0.f) * ir;
            cs += f * f;
        }
    }

    cs += __shfl_xor(cs, 16);                          // fold halves (M=v and M=v+8)
    if (lane < 16)
        atomicAdd(&cbuf[batch * NCELL + n0 + lane], cs);
}

// ---------------------------------------------------------------------------
// Kernel 5 (pass C): recompute tiles, apply both normalizations, geometric
// mask S, hinge loss, reduce to scalar.
// ---------------------------------------------------------------------------
__global__ void __launch_bounds__(256, 1)
k_loss(const _Float16* __restrict__ dn,
       const _Float16* __restrict__ wn,
       const float* __restrict__ invr,
       const float* __restrict__ invc,
       const float* __restrict__ warp,
       float* __restrict__ out) {
    int lane = threadIdx.x & 31;
    int wave = threadIdx.x >> 5;
    int m0   = blockIdx.x * 16;
    int batch = blockIdx.y;
    int hi   = lane >> 4;
    int lm   = lane & 15;

    const _Float16* arow = dn + (size_t)(batch * NCELL + m0 + lm) * DCH;
    v16h afrag[KCHUNKS];
    #pragma unroll
    for (int kc = 0; kc < KCHUNKS; ++kc)
        afrag[kc] = load_frag16x32(arow + kc * 32, hi);

    // per-row constants for this M-tile: inv row norm + warped center (row,col)
    float rl  = invr[batch * NCELL + m0 + lm];
    float wrl = warp[(batch * NCELL + m0 + lm) * 2 + 0];
    float wcl = warp[(batch * NCELL + m0 + lm) * 2 + 1];
    float irv[8], wrv[8], wcv[8];
    #pragma unroll
    for (int v = 0; v < 8; ++v) {
        int src = v + 8 * hi;
        irv[v] = __shfl(rl,  src);
        wrv[v] = __shfl(wrl, src);
        wcv[v] = __shfl(wcl, src);
    }

    float lsum = 0.f;
    const _Float16* wbase = wn + (size_t)batch * NCELL * DCH;

    for (int nt = wave; nt < TILES; nt += NWAVES) {
        int ng = nt * 16 + lm;                 // global column cell (this lane)
        const _Float16* brow = wbase + (size_t)ng * DCH;
        prefetch_row(wbase + (size_t)(clamp_tile(nt + NWAVES) * 16 + lm) * DCH);

        v16h bf[KCHUNKS];
        #pragma unroll
        for (int kc = 0; kc < KCHUNKS; ++kc)
            bf[kc] = load_frag16x32(brow + kc * 32, hi);

        __builtin_amdgcn_sched_barrier(0);   // all loads issue before any WMMA

        v8f acc0 = {}, acc1 = {};
        #pragma unroll
        for (int kc = 0; kc < KCHUNKS; kc += 2) {
            acc0 = __builtin_amdgcn_wmma_f32_16x16x32_f16(
                false, afrag[kc],     false, bf[kc],     (short)0, acc0, false, false);
            acc1 = __builtin_amdgcn_wmma_f32_16x16x32_f16(
                false, afrag[kc + 1], false, bf[kc + 1], (short)0, acc1, false, false);
        }

        float ic  = invc[batch * NCELL + ng];
        int krow  = ng / WCC;
        int kcol  = ng - krow * WCC;
        float cr  = krow * 8.0f + 4.0f;        // cell-center row
        float cc  = kcol * 8.0f + 4.0f;        // cell-center col
        #pragma unroll
        for (int v = 0; v < 8; ++v) {
            float val = fmaxf(acc0[v] + acc1[v], 0.f) * irv[v] * ic;  // normalized dot
            float dr = cr - wrv[v];
            float dc = cc - wcv[v];
            bool s = (dr * dr + dc * dc) <= DIST2;
            float pos = LAMBDA_D * fmaxf(POS_M - val, 0.f);
            float neg = fmaxf(val - NEG_M, 0.f);
            lsum += s ? pos : neg;
        }
    }

    lsum += __shfl_xor(lsum, 1);
    lsum += __shfl_xor(lsum, 2);
    lsum += __shfl_xor(lsum, 4);
    lsum += __shfl_xor(lsum, 8);
    lsum += __shfl_xor(lsum, 16);
    if (lane == 0) {
        const float inv_n2 = 1.0f / ((float)NCELL * (float)NCELL);
        atomicAdd(out, lsum * inv_n2);
    }
}

// ---------------------------------------------------------------------------
// Launch
// ---------------------------------------------------------------------------
extern "C" void kernel_launch(void* const* d_in, const int* in_sizes, int n_in,
                              void* d_out, int out_size, void* d_ws, size_t ws_size,
                              hipStream_t stream) {
    const float* desc  = (const float*)d_in[0];
    const float* wdesc = (const float*)d_in[1];
    const float* Hm    = (const float*)d_in[2];
    float* out = (float*)d_out;

    // workspace layout
    char* ws = (char*)d_ws;
    size_t szDesc = (size_t)BATCH * NCELL * DCH * sizeof(_Float16);   // 4.9 MB
    _Float16* dn   = (_Float16*)(ws);
    _Float16* wn   = (_Float16*)(ws + szDesc);
    float*    warp = (float*)(ws + 2 * szDesc);
    float*    rbuf = (float*)(ws + 2 * szDesc + (size_t)BATCH * NCELL * 2 * sizeof(float));
    float*    cbuf = (float*)(ws + 2 * szDesc + (size_t)BATCH * NCELL * 3 * sizeof(float));

    int total = BATCH * NCELL;
    int pblocks = (total + 255) / 256;

    k_prep<<<pblocks, 256, 0, stream>>>(desc, wdesc, Hm, dn, wn, warp, rbuf, cbuf, out);

    dim3 tg(TILES, BATCH);
    k_rownorm<<<tg, 256, 0, stream>>>(dn, wn, rbuf);
    k_finalize_inv<<<pblocks, 256, 0, stream>>>(rbuf, total);
    k_colnorm<<<tg, 256, 0, stream>>>(dn, wn, rbuf, cbuf);
    k_finalize_inv<<<pblocks, 256, 0, stream>>>(cbuf, total);
    k_loss<<<tg, 256, 0, stream>>>(dn, wn, rbuf, cbuf, warp, out);
}
